// TensorProductWeightsBlock_343597384376
// MI455X (gfx1250) — matrix operand order repaired and checked by reading, not compiled, not run on hardware
//
#include <hip/hip_runtime.h>

// einsum('be,ba,aek->bk') as GEMM: (B x 160) @ (160 x 128)
//   A-matrix rows (per edge): m[a*16+e] = attrs[b,a] * feats[b,e], built on the fly in bf16
//   B-matrix: weights reshaped (160 x 128), tiny (80KB) -> held in VGPRs per wave (bf16)
// Wave32 WMMA: v_wmma_f32_16x16x32_bf16, inner dim 160 = 5 chunks of K=32.
// Each wave: one 16-edge tile x one 64-column strip (two waves per tile cover K=128).
// Memory floor: ~616 MB moved @ 23.3 TB/s ~= 26 us; bf16 WMMA keeps compute well under that.
// __launch_bounds__(256, 4): <=256 VGPRs, 4 waves/SIMD. Ragged-tail handling is hoisted
// out of the hot loop so the allocator can keep all 20 B fragments register-resident.

#define A_DIM 10
#define E_DIM 16
#define K_DIM 128
#define KCHUNKS 5   // 160 / 32

typedef __attribute__((ext_vector_type(16))) __bf16 v16bf;
typedef __attribute__((ext_vector_type(8)))  float  v8f;

__global__ __launch_bounds__(256, 4) void mace_tpw_wmma_kernel(
    const float* __restrict__ attrs,   // (B, 10)
    const float* __restrict__ feats,   // (B, 16)
    const float* __restrict__ W,       // (10,16,128) viewed as (160,128) row-major
    float* __restrict__ out,           // (B, 128)
    int Bn, int npairs)
{
  const int tid  = blockIdx.x * blockDim.x + threadIdx.x;
  const int wave = tid >> 5;
  const int lane = tid & 31;
  const int half = wave & 1;           // 0 -> cols [0,64), 1 -> cols [64,128)
  const int pair = wave >> 1;          // tile sequence id

  const int l16 = lane & 15;           // row (A/C) or column (B/C) within tile
  const int hi  = lane >> 4;           // lane-half selector
  const int eoff = hi * 8;             // A frag: this lane-half covers e = eoff..eoff+7
  const int kln  = hi * 8;             // B frag: this lane-half covers K = kln..kln+7, kln+16..kln+23 (per chunk)
  const int colbase = half * 64;

  // ---- Preload B fragments (weights, bf16) into registers: 4 ntiles x 5 chunks ----
  // B layout (32x16, 16-bit): lane (0..15) = column N, element ei -> K = 32c + hi*8 + (ei & 7) + 16*(ei >> 3).
  v16bf Bfrag[4][KCHUNKS];
#pragma unroll
  for (int nl = 0; nl < 4; ++nl) {
    const int col = colbase + nl * 16 + l16;
#pragma unroll
    for (int c = 0; c < KCHUNKS; ++c) {
      const int kb = 32 * c + kln;
      v16bf b;
#pragma unroll
      for (int ei = 0; ei < 16; ++ei)
        b[ei] = (__bf16)W[(size_t)(kb + (ei & 7) + ((ei >> 3) << 4)) * K_DIM + col];
      Bfrag[nl][c] = b;
    }
  }

  const int nfull = Bn >> 4;           // number of full 16-edge tiles

  // ================= Hot loop: full tiles only, no guards =================
  for (int t = pair; t < nfull; t += npairs) {
    const int edge = t * 16 + l16;     // always < Bn here

    // Per-lane inputs: 8 edge feats (this lane-half of E) + all 10 node attrs.
    float4 f0 = *(const float4*)(feats + (size_t)edge * E_DIM + eoff);
    float4 f1 = *(const float4*)(feats + (size_t)edge * E_DIM + eoff + 4);
    float fl[8] = {f0.x, f0.y, f0.z, f0.w, f1.x, f1.y, f1.z, f1.w};
    float at[A_DIM];
#pragma unroll
    for (int i = 0; i < A_DIM / 2; ++i) {
      float2 a2 = *(const float2*)(attrs + (size_t)edge * A_DIM + 2 * i);
      at[2 * i] = a2.x;
      at[2 * i + 1] = a2.y;
    }

    // A frag element ei: a = 2c + ei/8, e = eoff + (ei & 7); A[m][k] = attrs[a]*feats[e].
    v16bf Afrag[KCHUNKS];
#pragma unroll
    for (int c = 0; c < KCHUNKS; ++c) {
      v16bf a;
#pragma unroll
      for (int ei = 0; ei < 16; ++ei)
        a[ei] = (__bf16)(at[2 * c + (ei >> 3)] * fl[ei & 7]);
      Afrag[c] = a;
    }

    // Single store base; all 32 stores use immediate offsets nl*64 + r*512 bytes.
    float* pbase = out + (size_t)(t * 16 + hi * 8) * K_DIM + colbase + l16;

#pragma unroll
    for (int nl = 0; nl < 4; ++nl) {
      v8f acc = {};
#pragma unroll
      for (int c = 0; c < KCHUNKS; ++c)
        acc = __builtin_amdgcn_wmma_f32_16x16x32_bf16(
            false, Afrag[c], false, Bfrag[nl][c], (short)0, acc, false, false);
#pragma unroll
      for (int r = 0; r < 8; ++r)
        pbase[(size_t)(r * K_DIM + nl * 16)] = acc[r];
    }
  }

  // ================= Ragged tail tile (at most one), cold path =================
  if ((Bn & 15) != 0) {
    const int t = nfull;
    // This wave-pair owns tile t iff t appears in its stride sequence.
    if (t >= pair && ((t - pair) % npairs) == 0) {
      const int edge = t * 16 + l16;
      const int eg   = edge < Bn ? edge : (Bn - 1);

      float4 f0 = *(const float4*)(feats + (size_t)eg * E_DIM + eoff);
      float4 f1 = *(const float4*)(feats + (size_t)eg * E_DIM + eoff + 4);
      float fl[8] = {f0.x, f0.y, f0.z, f0.w, f1.x, f1.y, f1.z, f1.w};
      float at[A_DIM];
#pragma unroll
      for (int i = 0; i < A_DIM / 2; ++i) {
        float2 a2 = *(const float2*)(attrs + (size_t)eg * A_DIM + 2 * i);
        at[2 * i] = a2.x;
        at[2 * i + 1] = a2.y;
      }

      v16bf Afrag[KCHUNKS];
#pragma unroll
      for (int c = 0; c < KCHUNKS; ++c) {
        v16bf a;
#pragma unroll
        for (int ei = 0; ei < 16; ++ei)
          a[ei] = (__bf16)(at[2 * c + (ei >> 3)] * fl[ei & 7]);
        Afrag[c] = a;
      }

#pragma unroll
      for (int nl = 0; nl < 4; ++nl) {
        v8f acc = {};
#pragma unroll
        for (int c = 0; c < KCHUNKS; ++c)
          acc = __builtin_amdgcn_wmma_f32_16x16x32_bf16(
              false, Afrag[c], false, Bfrag[nl][c], (short)0, acc, false, false);
        const int col = colbase + nl * 16 + l16;
#pragma unroll
        for (int r = 0; r < 8; ++r) {
          const int row = t * 16 + hi * 8 + r;
          if (row < Bn) out[(size_t)row * K_DIM + col] = acc[r];
        }
      }
    }
  }
}

extern "C" void kernel_launch(void* const* d_in, const int* in_sizes, int n_in,
                              void* d_out, int out_size, void* d_ws, size_t ws_size,
                              hipStream_t stream) {
  const float* attrs = (const float*)d_in[0];   // (B, 10)
  const float* feats = (const float*)d_in[1];   // (B, 16)
  const float* W     = (const float*)d_in[2];   // (10, 16, 128)
  float* out = (float*)d_out;

  const int Bn     = in_sizes[1] / E_DIM;       // number of edges
  const int ntiles = (Bn + 15) / 16;

  const int threads = 256;                      // 8 waves/block
  int blocks = (ntiles * 2 + 7) / 8;            // 2 waves per tile, 8 waves per block
  if (blocks > 1024) blocks = 1024;             // grid-stride beyond this
  if (blocks < 1) blocks = 1;
  const int npairs = (blocks * threads / 32) / 2;

  mace_tpw_wmma_kernel<<<blocks, threads, 0, stream>>>(
      attrs, feats, W, out, Bn, npairs);
}
